// Seq2Seq_85693187489990
// MI455X (gfx1250) — matrix operand order repaired and checked by reading, compiled, not run on hardware
//
#include <hip/hip_runtime.h>
#include <hip/hip_bf16.h>

#define B_TOT  16384
#define S_LEN  336
#define T_LEN  48
#define IN_DIM 5
#define H_DIM  64
#define G_DIM  256          // 4*H

// padded LDS leading dims (bank-conflict avoidance: 72*2B=144B -> 36-dword lane
// stride, 16 consecutive rows hit 16 distinct bank groups; 40*2B=80B -> 20 dwords, same)
#define LDW    72           // for [256][64] weights and [16][64] h stages
#define LDX    40           // for [256][<=32] input weights and [16][32] x stage

typedef __bf16 bf16_t;
typedef bf16_t v8bf  __attribute__((ext_vector_type(8)));
typedef bf16_t v16bf __attribute__((ext_vector_type(16)));
typedef float  v8f   __attribute__((ext_vector_type(8)));

union V16U { v16bf v; v8bf h[2]; };

// ---- LDS layout (dynamic shared) -------------------------------------------
#define OFF_WIH0   0                               // bf16 [256][LDX]  = 20480 B
#define OFF_WHH0   20480                           // bf16 [256][LDW]  = 36864 B
#define OFF_WIH1   57344
#define OFF_WHH1   94208
#define OFF_BIAS0  131072                          // f32 [256]
#define OFF_BIAS1  132096
#define OFF_FCW    133120                          // f32 [65] (W + b), pad to 272
#define OFF_STAGE  133392
// per-wave stage: h0 ping-pong (2*2304) + h1 ping-pong (2*2304) + x (1280)
#define WAVE_STAGE 10496
#define SH_BYTES   (OFF_STAGE + 8 * WAVE_STAGE)    // 217360 B  (< 320 KB/WGP)

static __device__ __forceinline__ float fast_sig(float x) {
  return 1.0f / (1.0f + __expf(-x));
}
static __device__ __forceinline__ float fast_tanh(float x) {
  float e = __expf(2.0f * x);
  return (e - 1.0f) / (e + 1.0f);
}

// A operand (16xK row-major bf16 stage -> ISA 16-bit A layout):
// lane L: M = L%16; lanes<16 take K = kbase+{0..7,16..23}, lanes>=16 K = kbase+{8..15,24..31}
static __device__ __forceinline__ v16bf load_A(const bf16_t* st, int ldk, int lane, int kbase) {
  int M  = lane & 15;
  int k0 = kbase + ((lane < 16) ? 0 : 8);
  V16U u;
  u.h[0] = *(const v8bf*)(st + M * ldk + k0);       // ds_load_b128
  u.h[1] = *(const v8bf*)(st + M * ldk + k0 + 16);  // ds_load_b128
  return u.v;
}

// B operand (W stored row-major [n][k], gemm uses W^T): lane L: N = L%16,
// lanes<16 cover K = kbase+{0..7,16..23}, lanes>=16 K = kbase+{8..15,24..31} (VGPR v packs 2v,2v+1)
static __device__ __forceinline__ v16bf load_B(const bf16_t* w, int ldk, int lane, int kbase, int nt) {
  int n  = nt * 16 + (lane & 15);
  int k0 = kbase + ((lane < 16) ? 0 : 8);
  V16U u;
  u.h[0] = *(const v8bf*)(w + n * ldk + k0);
  u.h[1] = *(const v8bf*)(w + n * ldk + k0 + 16);
  return u.v;
}

// One LSTM layer, gate-grouped: for each q in 0..3 accumulate only N-tiles
// {q, q+4, q+8, q+12} (i/f/g/o for hidden cols q*16..q*16+15) with v8f acc[4].
// The (NKCX+2)*4 B-tiles of the chain are software-pipelined (tile i+1's
// ds_load_b128 pair issues before the WMMA on tile i), first WMMA of each
// accumulator uses inline C=0, bias is folded into the activation, A-tiles are
// register-resident across all gate-groups, and cell state c stays in
// registers in exactly the WMMA C layout (no cross-lane traffic).
template<int NKCX>
static __device__ __forceinline__ void lstm_layer(
    const bf16_t* __restrict__ xa, int ldx, const bf16_t* __restrict__ Wx, int ldwx,
    const bf16_t* __restrict__ ha, const bf16_t* __restrict__ Wh,
    const float* __restrict__ bias, v8f c[4], bf16_t* __restrict__ hout, int lane) {
  int col   = lane & 15;
  int mbase = (lane >> 4) * 8;
  constexpr int NT = (NKCX + 2) * 4;

  v16bf ax[NKCX], ah[2];
#pragma unroll
  for (int kc = 0; kc < NKCX; ++kc) ax[kc] = load_A(xa, ldx, lane, kc * 32);
#pragma unroll
  for (int kc = 0; kc < 2; ++kc) ah[kc] = load_A(ha, LDW, lane, kc * 32);

#pragma unroll
  for (int q = 0; q < 4; ++q) {
    float bi = bias[(q)      * 16 + col];
    float bf = bias[(q + 4)  * 16 + col];
    float bg = bias[(q + 8)  * 16 + col];
    float bo = bias[(q + 12) * 16 + col];

    // tile i: segment seg = i/4 (x-chunks first, then h-chunks), gate g = i%4
    auto loadTile = [&](int i) -> v16bf {
      int seg = i >> 2, g = i & 3;
      const bf16_t* w = (seg < NKCX) ? Wx : Wh;
      int ldk = (seg < NKCX) ? ldwx : LDW;
      int kb  = ((seg < NKCX) ? seg : (seg - NKCX)) * 32;
      return load_B(w, ldk, lane, kb, q + 4 * g);
    };

    const v8f zero = {};
    v8f acc[4];
    v16bf bc = loadTile(0);
#pragma unroll
    for (int i = 0; i < NT; ++i) {
      v16bf bn;
      if (i + 1 < NT) bn = loadTile(i + 1);   // prefetch next tile before consuming current
      int seg = i >> 2, g = i & 3;
      v16bf a = (seg < NKCX) ? ax[seg] : ah[seg - NKCX];
      if (seg == 0)
        acc[g] = __builtin_amdgcn_wmma_f32_16x16x32_bf16(
            false, a, false, bc, (short)0, zero, false, false);   // inline C=0
      else
        acc[g] = __builtin_amdgcn_wmma_f32_16x16x32_bf16(
            false, a, false, bc, (short)0, acc[g], false, false);
      bc = bn;
    }

    // pointwise LSTM update for this gate-group (bias folded in here)
#pragma unroll
    for (int r = 0; r < 8; ++r) {
      float ig = fast_sig(acc[0][r] + bi);
      float fg = fast_sig(acc[1][r] + bf);
      float gg = fast_tanh(acc[2][r] + bg);
      float og = fast_sig(acc[3][r] + bo);
      float cn = fg * c[q][r] + ig * gg;
      c[q][r]  = cn;
      hout[(mbase + r) * LDW + q * 16 + col] = (bf16_t)(og * fast_tanh(cn));
    }
  }
}

// cooperative (256-thread) weight staging: f32 global -> bf16 LDS, zero-padded K
static __device__ void stage_weights(bf16_t* Wih0, bf16_t* Whh0, bf16_t* Wih1, bf16_t* Whh1,
                                     float* bias0, float* bias1,
                                     const float* gW0, int kin,
                                     const float* gU0, const float* gW1, const float* gU1,
                                     const float* b0i, const float* b0h,
                                     const float* b1i, const float* b1h, int tid) {
  for (int idx = tid; idx < G_DIM * LDX; idx += 256) {
    int n = idx / LDX, k = idx - n * LDX;
    Wih0[idx] = (k < kin) ? (bf16_t)gW0[n * kin + k] : (bf16_t)0.0f;
  }
  for (int idx = tid; idx < G_DIM * H_DIM; idx += 256) {
    int n = idx >> 6, k = idx & 63;
    Whh0[n * LDW + k] = (bf16_t)gU0[idx];
    Wih1[n * LDW + k] = (bf16_t)gW1[idx];
    Whh1[n * LDW + k] = (bf16_t)gU1[idx];
  }
  for (int n = tid; n < G_DIM; n += 256) {
    bias0[n] = b0i[n] + b0h[n];
    bias1[n] = b1i[n] + b1h[n];
  }
}

extern "C" __global__ __launch_bounds__(256)
void seq2seq_lstm_wmma(const float* __restrict__ src,
                       const float* eW0, const float* eU0, const float* eb0i, const float* eb0h,
                       const float* eW1, const float* eU1, const float* eb1i, const float* eb1h,
                       const float* dW0, const float* dU0, const float* db0i, const float* db0h,
                       const float* dW1, const float* dU1, const float* db1i, const float* db1h,
                       const float* fcW, const float* fcB,
                       float* __restrict__ out) {
  extern __shared__ __align__(16) char smem[];
  bf16_t* Wih0  = (bf16_t*)(smem + OFF_WIH0);
  bf16_t* Whh0  = (bf16_t*)(smem + OFF_WHH0);
  bf16_t* Wih1  = (bf16_t*)(smem + OFF_WIH1);
  bf16_t* Whh1  = (bf16_t*)(smem + OFF_WHH1);
  float*  bias0 = (float*) (smem + OFF_BIAS0);
  float*  bias1 = (float*) (smem + OFF_BIAS1);
  float*  fcw   = (float*) (smem + OFF_FCW);

  int tid  = threadIdx.x;
  int lane = tid & 31;
  int wave = tid >> 5;
  int col  = lane & 15;

  char* stg = smem + OFF_STAGE + wave * WAVE_STAGE;
  bf16_t* h0c = (bf16_t*)stg;            // ping
  bf16_t* h0n = (bf16_t*)(stg + 2304);   // pong
  bf16_t* h1c = (bf16_t*)(stg + 4608);
  bf16_t* h1n = (bf16_t*)(stg + 6912);
  bf16_t* xst = (bf16_t*)(stg + 9216);

  int b0 = blockIdx.x * 128 + wave * 16;
  const float* srow = src + (size_t)(b0 + col) * (S_LEN * IN_DIM);

  // ---- encoder weights into LDS, zero states -------------------------------
  stage_weights(Wih0, Whh0, Wih1, Whh1, bias0, bias1,
                eW0, IN_DIM, eU0, eW1, eU1, eb0i, eb0h, eb1i, eb1h, tid);
  for (int i = lane; i < 2 * 16 * LDW; i += 32) { h0c[i] = (bf16_t)0.0f; h1c[i] = (bf16_t)0.0f; }
  for (int i = lane; i < 16 * LDX; i += 32) xst[i] = (bf16_t)0.0f;
  __syncthreads();

  v8f c0[4], c1[4];
#pragma unroll
  for (int q = 0; q < 4; ++q)
#pragma unroll
    for (int r = 0; r < 8; ++r) { c0[q][r] = 0.0f; c1[q][r] = 0.0f; }

  // ---- encoder: 336 steps --------------------------------------------------
#pragma clang loop unroll(disable)
  for (int t = 0; t < S_LEN; ++t) {
    if (lane < 16) {
      const float* sp = srow + t * IN_DIM;
#pragma unroll
      for (int i = 0; i < IN_DIM; ++i) xst[col * LDX + i] = (bf16_t)sp[i];
      __builtin_prefetch(sp + 16 * IN_DIM, 0, 0);   // global_prefetch_b8 ahead of recurrence
    }
    lstm_layer<1>(xst, LDX, Wih0, LDX,    // x  @ Wih0^T (K 5->32)
                  h0c,      Whh0,         // h0 @ Whh0^T
                  bias0, c0, h0n, lane);
    lstm_layer<2>(h0n, LDW, Wih1, LDW,    // h0' @ Wih1^T
                  h1c,      Whh1,         // h1  @ Whh1^T
                  bias1, c1, h1n, lane);
    bf16_t* tp;
    tp = h0c; h0c = h0n; h0n = tp;
    tp = h1c; h1c = h1n; h1n = tp;
  }

  // ---- phase switch: decoder weights overwrite encoder weights -------------
  __syncthreads();
  stage_weights(Wih0, Whh0, Wih1, Whh1, bias0, bias1,
                dW0, 1, dU0, dW1, dU1, db0i, db0h, db1i, db1h, tid);
  for (int i = tid; i < 64; i += 256) fcw[i] = fcW[i];
  if (tid == 0) fcw[64] = fcB[0];
  if (lane < 16) {
#pragma unroll
    for (int k = 1; k < IN_DIM; ++k) xst[col * LDX + k] = (bf16_t)0.0f;   // clear enc inputs
    xst[col * LDX] = (bf16_t)srow[(S_LEN - 1) * IN_DIM + 3];              // dec_in0 = src[:,-1,3]
  }
  __syncthreads();

  // ---- decoder: 48 steps with scalar feedback ------------------------------
#pragma clang loop unroll(disable)
  for (int t = 0; t < T_LEN; ++t) {
    lstm_layer<1>(xst, LDX, Wih0, LDX,    // x(scalar, K 1->32)
                  h0c,      Whh0,
                  bias0, c0, h0n, lane);
    lstm_layer<2>(h0n, LDW, Wih1, LDW,
                  h1c,      Whh1,
                  bias1, c1, h1n, lane);
    if (lane < 16) {
      float p = fcw[64];
#pragma unroll 8
      for (int j = 0; j < H_DIM; ++j) p += (float)h1n[col * LDW + j] * fcw[j];
      out[(size_t)(b0 + col) * T_LEN + t] = p;
      xst[col * LDX] = (bf16_t)p;                 // feedback
    }
    bf16_t* tp;
    tp = h0c; h0c = h0n; h0n = tp;
    tp = h1c; h1c = h1n; h1n = tp;
  }
}

extern "C" void kernel_launch(void* const* d_in, const int* in_sizes, int n_in,
                              void* d_out, int out_size, void* d_ws, size_t ws_size,
                              hipStream_t stream) {
  (void)in_sizes; (void)n_in; (void)out_size; (void)d_ws; (void)ws_size;
  const float* src  = (const float*)d_in[0];
  const float* eW0  = (const float*)d_in[1];
  const float* eU0  = (const float*)d_in[2];
  const float* eb0i = (const float*)d_in[3];
  const float* eb0h = (const float*)d_in[4];
  const float* eW1  = (const float*)d_in[5];
  const float* eU1  = (const float*)d_in[6];
  const float* eb1i = (const float*)d_in[7];
  const float* eb1h = (const float*)d_in[8];
  const float* dW0  = (const float*)d_in[9];
  const float* dU0  = (const float*)d_in[10];
  const float* db0i = (const float*)d_in[11];
  const float* db0h = (const float*)d_in[12];
  const float* dW1  = (const float*)d_in[13];
  const float* dU1  = (const float*)d_in[14];
  const float* db1i = (const float*)d_in[15];
  const float* db1h = (const float*)d_in[16];
  const float* fcW  = (const float*)d_in[17];
  const float* fcB  = (const float*)d_in[18];
  float* out = (float*)d_out;

  (void)hipFuncSetAttribute((const void*)seq2seq_lstm_wmma,
                            hipFuncAttributeMaxDynamicSharedMemorySize, SH_BYTES);

  dim3 grid(B_TOT / 128), block(256);  // 8 waves * 16 batch rows each
  seq2seq_lstm_wmma<<<grid, block, SH_BYTES, stream>>>(
      src, eW0, eU0, eb0i, eb0h, eW1, eU1, eb1i, eb1h,
      dW0, dU0, db0i, db0h, dW1, dU1, db1i, db1h, fcW, fcB, out);
}
